// DeformableTemporalSelfAttention_35570919145705
// MI455X (gfx1250) — matrix operand order repaired and checked by reading, not compiled, not run on hardware
//
#include <hip/hip_runtime.h>

// ---------------- problem constants ----------------
#define BATCH   8
#define SLEN    2048
#define DMODEL  1024
#define HEADS   16
#define NPT     8
#define DHEAD   64
#define DFEAT   256
#define ROWS    (BATCH * SLEN)       // 16384
#define KCTX    (DMODEL + 2 * DFEAT) // 1536
#define HP      (HEADS * NPT)        // 128

typedef __bf16 bf16;
typedef bf16  v16bf __attribute__((ext_vector_type(16)));
typedef bf16  v8bf  __attribute__((ext_vector_type(8)));
typedef float v8f   __attribute__((ext_vector_type(8)));

__device__ __constant__ float c_anchors[NPT] = {
    0.0f, 1.0022248f, 3.0089043f, 7.0267273f,
    15.0713135f, 31.1783776f, 63.4283523f, 128.0f};
__device__ __constant__ float c_bounds[8] = {
    0.0f, 0.5f, 1.0f, 2.0f, 4.0f, 8.0f, 12.0f, 24.0f};

__device__ __forceinline__ bf16 f2bf(float f) {
  union { float f; unsigned u; } a; a.f = f;
  unsigned u = a.u;
  u += 0x7FFFu + ((u >> 16) & 1u);          // round-to-nearest-even
  union { unsigned short s; bf16 b; } r;
  r.s = (unsigned short)(u >> 16);
  return r.b;
}

__device__ __forceinline__ void sched_fence() {
#if __has_builtin(__builtin_amdgcn_sched_barrier)
  __builtin_amdgcn_sched_barrier(0);   // nothing may cross
#endif
}

// ---------------- per-token features: build bf16 concat input ----------------
// Abuf[row][0..1023] = x ; [1024..1279] = cont_feat ; [1280..1535] = bucket_feat
__global__ __launch_bounds__(256) void feat_kernel(
    const float* __restrict__ x, const float* __restrict__ td,
    const float* __restrict__ Wc, const float* __restrict__ bc,
    const float* __restrict__ Eb,
    bf16* __restrict__ Abuf, float* __restrict__ dtbuf) {
  const int row = blockIdx.x;
  const int t = threadIdx.x;
  const float dtv = fmaxf(td[row], 0.0f);
  if (t == 0) dtbuf[row] = dtv;
  const float dl = log1pf(dtv);
  int bucket = 0;
#pragma unroll
  for (int i = 0; i < 8; ++i) bucket += (c_bounds[i] < dtv) ? 1 : 0;
  bf16* arow = Abuf + (size_t)row * KCTX;
  const float* xrow = x + (size_t)row * DMODEL;
#pragma unroll
  for (int i = 0; i < 4; ++i) {
    int j = t + i * 256;
    arow[j] = f2bf(xrow[j]);
  }
  arow[DMODEL + t]          = f2bf(dl * Wc[t] + bc[t]);
  arow[DMODEL + DFEAT + t]  = f2bf(Eb[bucket * DFEAT + t]);
}

// ---------------- per-batch inclusive scan of dt -> elapsed ----------------
__global__ __launch_bounds__(256) void cumsum_kernel(
    const float* __restrict__ dtb, float* __restrict__ el) {
  __shared__ float tot[256];
  const int b = blockIdx.x, t = threadIdx.x;
  const float* in = dtb + (size_t)b * SLEN;
  float* out = el + (size_t)b * SLEN;
  float v[8]; float s = 0.0f;
#pragma unroll
  for (int i = 0; i < 8; ++i) { v[i] = in[t * 8 + i]; s += v[i]; }
  tot[t] = s;
  __syncthreads();
  for (int off = 1; off < 256; off <<= 1) {
    float add = (t >= off) ? tot[t - off] : 0.0f;
    __syncthreads();
    tot[t] += add;
    __syncthreads();
  }
  float acc = (t > 0) ? tot[t - 1] : 0.0f;
#pragma unroll
  for (int i = 0; i < 8; ++i) { acc += v[i]; out[t * 8 + i] = acc; }
}

// ---------------- weight fp32[K,N] -> bf16 transposed [N,K] ----------------
__global__ __launch_bounds__(256) void wconv_kernel(
    const float* __restrict__ src, bf16* __restrict__ dst, int K, int N) {
  int idx = blockIdx.x * 256 + threadIdx.x;
  if (idx >= K * N) return;
  int k = idx / N, n = idx - k * N;
  dst[(size_t)n * K + k] = f2bf(src[idx]);
}

// ---------------- bf16 WMMA GEMM: C[M,N] = act(A[M,K] @ Bt^T + bias) ---------
// Bt is [N,K] row-major. Block = 8 waves; tile 256(M) x 64(N); wave = 32x64.
// Double-buffered LDS B-staging (register pass-through), 1 barrier / K-step.
// Per step: all 8 LDS fragment loads issued first (sched_barrier), then
// 8 back-to-back WMMAs on disjoint registers (no intra-burst waits).
template <bool OUT_BF16, bool ACT_TANH>
__global__ __launch_bounds__(256) void gemm_bf16_wmma(
    const bf16* __restrict__ A, int lda,
    const bf16* __restrict__ Bt,
    const float* __restrict__ bias,
    void* __restrict__ Cv, int ldc, int K) {
  __shared__ __align__(32) bf16 Bs[2][64 * 32];
  const int tid  = threadIdx.x;
  const int wave = tid >> 5;
  const int lane = tid & 31;
  const int half = lane >> 4;   // which 16-lane half
  const int l16  = lane & 15;

  const int rowA0   = blockIdx.x * 256 + wave * 32 + l16;   // A frag 0 rows
  const int colBase = blockIdx.y * 64;
  const bf16* Arow0 = A + (size_t)rowA0 * lda;
  const bf16* Arow1 = Arow0 + (size_t)16 * lda;             // A frag 1 rows

  // B staging: thread -> (n = tid/4 in [0,64), 8 contiguous k at (tid%4)*8)
  const int sn = tid >> 2;
  const int sk = (tid & 3) * 8;
  const bf16* Bg = Bt + (size_t)(colBase + sn) * K + sk;
  const int stageOff = sn * 32 + sk;

  v8f acc[8] = {};
  const int kb0 = half * 8;   // A K-layout: lanes<16: {0..7,16..23}; >=16: {8..15,24..31}

  const int nsteps = K >> 5;
  v8bf rstage = *(const v8bf*)Bg;   // preload tile 0
  for (int it = 0; it < nsteps; ++it) {
    *(v8bf*)&Bs[it & 1][stageOff] = rstage;
    if (it + 1 < nsteps) {
      rstage = *(const v8bf*)(Bg + (it + 1) * 32);
      __builtin_prefetch(Bg + (it + 2) * 32, 0, 1);   // global_prefetch_b8
    }
    __syncthreads();   // Bs[it&1] ready; prior reads of it&1 retired 2 iters ago

    const int k0 = it * 32;
    v8bf a00 = *(const v8bf*)(Arow0 + k0 + kb0);
    v8bf a01 = *(const v8bf*)(Arow0 + k0 + kb0 + 16);
    v8bf a10 = *(const v8bf*)(Arow1 + k0 + kb0);
    v8bf a11 = *(const v8bf*)(Arow1 + k0 + kb0 + 16);
    v16bf aLo, aHi;
#pragma unroll
    for (int i = 0; i < 8; ++i) {
      aLo[i] = a00[i]; aLo[i + 8] = a01[i];
      aHi[i] = a10[i]; aHi[i + 8] = a11[i];
    }

    // Load ALL B fragments first, then fence the scheduler, then WMMA burst.
    // B layout: lane = col, lanes<16 -> K {0..7,16..23} ; lanes>=16 -> K {8..15,24..31}
    const bf16* bsr = &Bs[it & 1][l16 * 32 + half * 8];
    v16bf bfrag[4];
#pragma unroll
    for (int nt = 0; nt < 4; ++nt) {
      v8bf b0 = *(const v8bf*)(bsr + nt * 16 * 32);
      v8bf b1 = *(const v8bf*)(bsr + nt * 16 * 32 + 16);
      for (int i = 0; i < 8; ++i) { bfrag[nt][i] = b0[i]; bfrag[nt][i + 8] = b1[i]; }
    }

    sched_fence();   // keep all 4 bfrags live; single dscnt wait before burst

#pragma unroll
    for (int nt = 0; nt < 4; ++nt) {
      acc[2 * nt + 0] = __builtin_amdgcn_wmma_f32_16x16x32_bf16(
          false, aLo, false, bfrag[nt], (short)0, acc[2 * nt + 0], false, false);
      acc[2 * nt + 1] = __builtin_amdgcn_wmma_f32_16x16x32_bf16(
          false, aHi, false, bfrag[nt], (short)0, acc[2 * nt + 1], false, false);
    }
  }

  // D layout: VGPR r -> row = r + 8*(lane/16); col = lane%16
  const int rowD0 = blockIdx.x * 256 + wave * 32 + half * 8;
#pragma unroll
  for (int nt = 0; nt < 4; ++nt) {
    const int col = colBase + nt * 16 + l16;
    const float bcol = bias[col];
#pragma unroll
    for (int mt = 0; mt < 2; ++mt) {
      const int rowD = rowD0 + mt * 16;
#pragma unroll
      for (int r2 = 0; r2 < 8; ++r2) {
        float vv = acc[2 * nt + mt][r2] + bcol;
        if (ACT_TANH) vv = tanhf(vv);
        size_t ro = (size_t)(rowD + r2) * ldc + col;
        if (OUT_BF16) ((bf16*)Cv)[ro] = f2bf(vv);
        else          ((float*)Cv)[ro] = vv;
      }
    }
  }
}

// ---------------- deformable attention: one wave32 per (b,h,s) --------------
__global__ __launch_bounds__(256) void attn_kernel(
    const float* __restrict__ q, const float* __restrict__ k,
    const float* __restrict__ v, const float* __restrict__ offb,
    const float* __restrict__ psb, const float* __restrict__ elapsed,
    const float* __restrict__ pbias, const float* __restrict__ tdw,
    const float* __restrict__ tdb, bf16* __restrict__ attnb) {
  const int wid  = (blockIdx.x * blockDim.x + threadIdx.x) >> 5;
  const int lane = threadIdx.x & 31;
  const int s = wid & (SLEN - 1);
  const int h = (wid >> 11) & (HEADS - 1);
  const int b = wid >> 15;

  const size_t rowT = (size_t)b * SLEN + s;
  const int dbase = h * DHEAD + lane * 2;   // each lane owns 2 head dims
  const float* qp = q + rowT * DMODEL + dbase;
  const float q0 = qp[0], q1 = qp[1];
  const float eq = elapsed[rowT];

  float sv0[NPT], sv1[NPT], sc[NPT];
#pragma unroll
  for (int p = 0; p < NPT; ++p) {
    const float off = offb[rowT * HP + h * NPT + p];
    float pos = (float)s - c_anchors[p] + off * 8.0f;
    pos = fminf(fmaxf(pos, 0.0f), (float)s);
    int left = (int)floorf(pos);
    left = left < 0 ? 0 : (left > SLEN - 1 ? SLEN - 1 : left);
    int right = left + 1 > SLEN - 1 ? SLEN - 1 : left + 1;
    const float alpha = pos - (float)left;

    const size_t rl = (size_t)b * SLEN + left;
    const size_t rr = (size_t)b * SLEN + right;
    const float* kl = k + rl * DMODEL + dbase;
    const float* kr = k + rr * DMODEL + dbase;
    const float* vl = v + rl * DMODEL + dbase;
    const float* vr = v + rr * DMODEL + dbase;

    const float sk0 = kl[0] + (kr[0] - kl[0]) * alpha;
    const float sk1 = kl[1] + (kr[1] - kl[1]) * alpha;
    sv0[p] = vl[0] + (vr[0] - vl[0]) * alpha;
    sv1[p] = vl[1] + (vr[1] - vl[1]) * alpha;

    float dp = q0 * sk0 + q1 * sk1;
#pragma unroll
    for (int m = 16; m >= 1; m >>= 1) dp += __shfl_xor(dp, m, 32);

    const float el = elapsed[rl];
    const float se = el + (elapsed[rr] - el) * alpha;
    const float rel = log1pf(fmaxf(eq - se, 0.0f));
    const float w = tdw[h * NPT + p];
    const float decay = (w > 20.0f) ? w : log1pf(expf(w));   // softplus
    sc[p] = dp * 0.125f + psb[rowT * HP + h * NPT + p] +
            tdb[h * NPT + p] - decay * rel + pbias[h * NPT + p];
  }

  float m = sc[0];
#pragma unroll
  for (int p = 1; p < NPT; ++p) m = fmaxf(m, sc[p]);
  float e[NPT]; float sum = 0.0f;
#pragma unroll
  for (int p = 0; p < NPT; ++p) { e[p] = expf(sc[p] - m); sum += e[p]; }
  const float inv = 1.0f / sum;
  float o0 = 0.0f, o1 = 0.0f;
#pragma unroll
  for (int p = 0; p < NPT; ++p) { o0 += e[p] * sv0[p]; o1 += e[p] * sv1[p]; }
  o0 *= inv; o1 *= inv;

  // packed 2 x bf16 store (4B aligned: dbase is even)
  union { unsigned u; bf16 b2[2]; } st;
  st.b2[0] = f2bf(o0);
  st.b2[1] = f2bf(o1);
  *(unsigned*)(attnb + rowT * DMODEL + dbase) = st.u;
}

// ---------------- launcher ----------------
extern "C" void kernel_launch(void* const* d_in, const int* in_sizes, int n_in,
                              void* d_out, int out_size, void* d_ws,
                              size_t ws_size, hipStream_t stream) {
  const float* x    = (const float*)d_in[0];
  const float* td   = (const float*)d_in[1];
  const float* Wq   = (const float*)d_in[2];
  const float* bq   = (const float*)d_in[3];
  const float* Wk   = (const float*)d_in[4];
  const float* bk   = (const float*)d_in[5];
  const float* Wv   = (const float*)d_in[6];
  const float* bv   = (const float*)d_in[7];
  const float* Wc   = (const float*)d_in[8];
  const float* bc   = (const float*)d_in[9];
  const float* Eb   = (const float*)d_in[10];
  const float* Wt   = (const float*)d_in[11];
  const float* bt   = (const float*)d_in[12];
  const float* Woff = (const float*)d_in[13];
  const float* boff = (const float*)d_in[14];
  const float* Wps  = (const float*)d_in[15];
  const float* bps  = (const float*)d_in[16];
  const float* Wo   = (const float*)d_in[17];
  const float* bo   = (const float*)d_in[18];
  const float* pb   = (const float*)d_in[19];
  const float* tdw  = (const float*)d_in[20];
  const float* tdb  = (const float*)d_in[21];

  char* p = (char*)d_ws;
  auto carve = [&](size_t bytes) {
    char* r = p;
    p += (bytes + 255) & ~(size_t)255;
    return r;
  };
  bf16*  Abuf   = (bf16*)carve((size_t)ROWS * KCTX * 2);
  float* qbuf   = (float*)carve((size_t)ROWS * DMODEL * 4);
  float* kbuf   = (float*)carve((size_t)ROWS * DMODEL * 4);
  float* vbuf   = (float*)carve((size_t)ROWS * DMODEL * 4);
  bf16*  ctxb   = (bf16*)carve((size_t)ROWS * DMODEL * 2);
  float* offbuf = (float*)carve((size_t)ROWS * HP * 4);
  float* psbuf  = (float*)carve((size_t)ROWS * HP * 4);
  bf16*  attnb  = (bf16*)carve((size_t)ROWS * DMODEL * 2);
  float* dtbuf  = (float*)carve((size_t)ROWS * 4);
  float* elbuf  = (float*)carve((size_t)ROWS * 4);
  bf16*  WqT    = (bf16*)carve((size_t)DMODEL * DMODEL * 2);
  bf16*  WkT    = (bf16*)carve((size_t)DMODEL * DMODEL * 2);
  bf16*  WvT    = (bf16*)carve((size_t)DMODEL * DMODEL * 2);
  bf16*  WoT    = (bf16*)carve((size_t)DMODEL * DMODEL * 2);
  bf16*  WtT    = (bf16*)carve((size_t)KCTX * DMODEL * 2);
  bf16*  WoffT  = (bf16*)carve((size_t)DMODEL * HP * 2);
  bf16*  WpsT   = (bf16*)carve((size_t)DMODEL * HP * 2);

  // 1) features + bf16 concat input
  feat_kernel<<<ROWS, 256, 0, stream>>>(x, td, Wc, bc, Eb, Abuf, dtbuf);
  // 2) elapsed = cumsum(dt)
  cumsum_kernel<<<BATCH, 256, 0, stream>>>(dtbuf, elbuf);
  // 3) weights -> bf16 transposed
  auto wc = [&](const float* src, bf16* dst, int K, int N) {
    int n = K * N;
    wconv_kernel<<<(n + 255) / 256, 256, 0, stream>>>(src, dst, K, N);
  };
  wc(Wq, WqT, DMODEL, DMODEL);
  wc(Wk, WkT, DMODEL, DMODEL);
  wc(Wv, WvT, DMODEL, DMODEL);
  wc(Wo, WoT, DMODEL, DMODEL);
  wc(Wt, WtT, KCTX, DMODEL);
  wc(Woff, WoffT, DMODEL, HP);
  wc(Wps, WpsT, DMODEL, HP);

  // 4) WMMA GEMMs (block tile 256x64)
  const dim3 blk(256);
  const dim3 gD(ROWS / 256, DMODEL / 64);
  const dim3 gH(ROWS / 256, HP / 64);
  gemm_bf16_wmma<false, false><<<gD, blk, 0, stream>>>(Abuf, KCTX, WqT, bq,
                                                       qbuf, DMODEL, DMODEL);
  gemm_bf16_wmma<false, false><<<gD, blk, 0, stream>>>(Abuf, KCTX, WkT, bk,
                                                       kbuf, DMODEL, DMODEL);
  gemm_bf16_wmma<false, false><<<gD, blk, 0, stream>>>(Abuf, KCTX, WvT, bv,
                                                       vbuf, DMODEL, DMODEL);
  gemm_bf16_wmma<true, false><<<gD, blk, 0, stream>>>(Abuf, KCTX, WtT, bt,
                                                      ctxb, DMODEL, KCTX);
  gemm_bf16_wmma<false, true><<<gH, blk, 0, stream>>>(ctxb, DMODEL, WoffT, boff,
                                                      offbuf, HP, DMODEL);
  gemm_bf16_wmma<false, false><<<gH, blk, 0, stream>>>(ctxb, DMODEL, WpsT, bps,
                                                       psbuf, HP, DMODEL);

  // 5) deformable attention (wave32 per (b,h,s))
  const int nWaves = BATCH * HEADS * SLEN;             // 262144
  attn_kernel<<<nWaves / 8, 256, 0, stream>>>(qbuf, kbuf, vbuf, offbuf, psbuf,
                                              elbuf, pb, tdw, tdb, attnb);

  // 6) output projection -> d_out (fp32)
  gemm_bf16_wmma<false, false><<<gD, blk, 0, stream>>>(
      attnb, DMODEL, WoT, bo, (float*)d_out, DMODEL, DMODEL);
}
